// MoELayer_10917806866937
// MI455X (gfx1250) — matrix operand order repaired and checked by reading, not compiled, not run on hardware
//
#include <hip/hip_runtime.h>
#include <hip/hip_bf16.h>
#include <math.h>

// Problem constants (B=2, S=2048, D=1024, E=8, H=4096, K=2)
#define TT 4096
#define DD 1024
#define EE 8
#define HH 4096
#define TOPK 2

typedef __attribute__((ext_vector_type(16))) __bf16 v16bf;
typedef __attribute__((ext_vector_type(8)))  __bf16 v8bf;
typedef __attribute__((ext_vector_type(8)))  float  v8f;
typedef __attribute__((ext_vector_type(4)))  unsigned int v4u;
typedef __attribute__((ext_vector_type(8)))  int v8i_;
typedef __attribute__((ext_vector_type(4)))  int v4i_;

union ABReg {
    v16bf v;
    struct { v8bf lo, hi; } h;
};

// ---------------------------------------------------------------------------
// Tensor Data Mover descriptor (D#) per CDNA5 ISA 8.3/8.4 and issue helper.
// ---------------------------------------------------------------------------
__device__ __forceinline__ void tdm_load(unsigned int lds_off, unsigned long long gaddr,
                                         unsigned int tile_w, unsigned int tile_h,
                                         unsigned long long row_stride_elems,
                                         unsigned int td0, unsigned int td1,
                                         bool gather, v4i_ g2, v4i_ g3) {
    v4u g0;
    g0[0] = 1u | (gather ? (1u << 31) : 0u);                      // count=1, gather_mode, 16-bit idx
    g0[1] = lds_off;                                              // LDS byte address
    g0[2] = (unsigned int)gaddr;                                  // global_addr[31:0]
    g0[3] = ((unsigned int)(gaddr >> 32) & 0x1FFFFFFu) | 0x80000000u; // addr[56:32] | type=2
    v8i_ g1;
    g1[0] = (1 << 16);                                            // wg_mask=0, data_size=1 (2 bytes)
    g1[1] = (int)((td0 & 0xFFFFu) << 16);                         // tensor_dim0[15:0]
    g1[2] = (int)((td0 >> 16) | ((td1 & 0xFFFFu) << 16));         // dim0[31:16], dim1[15:0]
    g1[3] = (int)((td1 >> 16) | (tile_w << 16));                  // dim1[31:16], tile_dim0
    g1[4] = (int)(tile_h & 0xFFFFu);                              // tile_dim1, tile_dim2=0
    g1[5] = (int)(unsigned int)row_stride_elems;                  // tensor_dim0_stride[31:0]
    g1[6] = (int)((unsigned int)(row_stride_elems >> 32) & 0xFFFFu);
    g1[7] = 0;
#if __clang_major__ >= 23
    v8i_ z8 = {0, 0, 0, 0, 0, 0, 0, 0};
    __builtin_amdgcn_tensor_load_to_lds(g0, g1, g2, g3, z8, 0);
#else
    __builtin_amdgcn_tensor_load_to_lds(g0, g1, g2, g3, 0);
#endif
}

__device__ __forceinline__ unsigned int lds_off_of(const void* p) {
    return (unsigned int)(size_t)p;   // LDS aperture: addr[31:0] == LDS byte offset
}

// ---------------------------------------------------------------------------
// k_init: zero final-output accumulator region and per-expert counters
// ---------------------------------------------------------------------------
__global__ __launch_bounds__(256) void k_init(float* __restrict__ out, int* __restrict__ cnt) {
    size_t i = (size_t)blockIdx.x * blockDim.x + threadIdx.x;
    float4 z = make_float4(0.f, 0.f, 0.f, 0.f);
    size_t n4 = (size_t)TT * DD / 4;
    if (i < n4) ((float4*)out)[i] = z;
    if (blockIdx.x == 0 && threadIdx.x < EE) cnt[threadIdx.x] = 0;
}

// ---------------------------------------------------------------------------
// k_transpose_cast: [E][R][C] f32  ->  [E][C][R] bf16  (64x64 LDS tiles)
// ---------------------------------------------------------------------------
__global__ __launch_bounds__(256) void k_transpose_cast(const float* __restrict__ src,
                                                        __bf16* __restrict__ dst,
                                                        int R, int C) {
    __shared__ __bf16 t[64][65];
    int e = blockIdx.z;
    int r0 = blockIdx.y * 64, c0 = blockIdx.x * 64;
    int tid = threadIdx.x;
#pragma unroll 4
    for (int i = 0; i < 16; ++i) {
        int idx = tid + 256 * i;
        int rr = idx >> 6, cc = idx & 63;
        t[cc][rr] = (__bf16)src[((size_t)e * R + r0 + rr) * C + c0 + cc];
    }
    __builtin_prefetch(&src[((size_t)e * R + r0) * C + c0 + 64], 0, 1);
    __syncthreads();
#pragma unroll 4
    for (int i = 0; i < 16; ++i) {
        int idx = tid + 256 * i;
        int cc = idx >> 6, rr = idx & 63;
        dst[((size_t)e * C + c0 + cc) * R + r0 + rr] = t[cc][rr];
    }
}

// ---------------------------------------------------------------------------
// k_gate: one wave per token. noisy top-2 gate, dispatch lists, bf16 x copy.
// ---------------------------------------------------------------------------
__global__ __launch_bounds__(256) void k_gate(const float* __restrict__ x,
                                              const float* __restrict__ noise,
                                              const float* __restrict__ Wg,
                                              const float* __restrict__ nw,
                                              __bf16* __restrict__ xb,
                                              int* __restrict__ pair_tok,
                                              float* __restrict__ pair_gate,
                                              int* __restrict__ cnt,
                                              int* __restrict__ idx_out) {
    int lane = threadIdx.x & 31;
    int wave = threadIdx.x >> 5;
    int t = blockIdx.x * 8 + wave;
    if (t >= TT) return;

    float acc[EE];
#pragma unroll
    for (int e = 0; e < EE; ++e) acc[e] = 0.f;

#pragma unroll 4
    for (int j = 0; j < DD / 32; ++j) {
        int d = lane + 32 * j;
        float xv = x[(size_t)t * DD + d];
        xb[(size_t)t * DD + d] = (__bf16)xv;
#pragma unroll
        for (int e = 0; e < EE; ++e) acc[e] += xv * Wg[e * DD + d];
    }
#pragma unroll
    for (int m = 16; m >= 1; m >>= 1) {
#pragma unroll
        for (int e = 0; e < EE; ++e) acc[e] += __shfl_xor(acc[e], m, 32);
    }

    if (lane == 0) {
        float nl[EE];
#pragma unroll
        for (int e = 0; e < EE; ++e)
            nl[e] = acc[e] + noise[(size_t)t * EE + e] * nw[e];
        int i1 = 0; float v1 = nl[0];
#pragma unroll
        for (int e = 1; e < EE; ++e) if (nl[e] > v1) { v1 = nl[e]; i1 = e; }
        int i2 = (i1 == 0) ? 1 : 0; float v2 = nl[i2];
#pragma unroll
        for (int e = 0; e < EE; ++e)
            if (e != i1 && nl[e] > v2) { v2 = nl[e]; i2 = e; }
        float g1 = 1.f / (1.f + __expf(v2 - v1));
        float g2 = 1.f - g1;
        idx_out[t * TOPK + 0] = i1;
        idx_out[t * TOPK + 1] = i2;
        int s1 = atomicAdd(&cnt[i1], 1);
        pair_tok[i1 * TT + s1]  = t;
        pair_gate[i1 * TT + s1] = g1;
        int s2 = atomicAdd(&cnt[i2], 1);
        pair_tok[i2 * TT + s2]  = t;
        pair_gate[i2 * TT + s2] = g2;
    }
}

// ---------------------------------------------------------------------------
// k_off: CSR offsets over expert counts
// ---------------------------------------------------------------------------
__global__ void k_off(const int* __restrict__ cnt, int* __restrict__ off) {
    if (threadIdx.x == 0 && blockIdx.x == 0) {
        int a = 0;
        off[0] = 0;
        for (int e = 0; e < EE; ++e) { a += cnt[e]; off[e + 1] = a; }
    }
}

// ---------------------------------------------------------------------------
// k1_up: stage-1 grouped GEMM + SwiGLU.  Block tile 32(M) x 128(N), K over D.
// TDM gather for token rows, TDM 2D tiles for weights, double-buffered LDS.
// Register blocking: wave = N-strip; 4 accumulators (2 M-subtiles x W1/W2);
// per 32-wide k step 4 LDS operand fetches feed 4 WMMAs.
// ---------------------------------------------------------------------------
#define NK1 (DD / 64)
__global__ __launch_bounds__(256) void k1_up(const __bf16* __restrict__ w1t,
                                             const __bf16* __restrict__ w2t,
                                             const float* __restrict__ b1,
                                             const float* __restrict__ b2,
                                             const __bf16* __restrict__ xb,
                                             const int* __restrict__ pair_tok,
                                             const int* __restrict__ cnt,
                                             const int* __restrict__ off,
                                             __bf16* __restrict__ hh) {
    int e = blockIdx.z, mt = blockIdx.y, nt = blockIdx.x;   // nt over HH/128
    int ce = cnt[e];
    if (mt * 32 >= ce) return;

    __shared__ __bf16 xs[2][32][64];
    __shared__ __bf16 w1s[2][128][64];
    __shared__ __bf16 w2s[2][128][64];
    __shared__ int tokL[32];

    int tid = threadIdx.x;
    int lane = tid & 31, wave = tid >> 5;
    if (tid < 32) {
        int slot = mt * 32 + tid;
        if (slot >= ce) slot = ce - 1;
        tokL[tid] = pair_tok[e * TT + slot];
    }
    __syncthreads();

    v4i_ gA0, gA1, gB0, gB1;
#pragma unroll
    for (int i = 0; i < 4; ++i) {
        gA0[i] = (tokL[2 * i]      & 0xFFFF) | (tokL[2 * i + 1]  << 16);
        gA1[i] = (tokL[8 + 2 * i]  & 0xFFFF) | (tokL[9 + 2 * i]  << 16);
        gB0[i] = (tokL[16 + 2 * i] & 0xFFFF) | (tokL[17 + 2 * i] << 16);
        gB1[i] = (tokL[24 + 2 * i] & 0xFFFF) | (tokL[25 + 2 * i] << 16);
    }
    v4i_ z4 = {0, 0, 0, 0};

    unsigned long long xbase  = (unsigned long long)(size_t)xb;
    unsigned long long w1base = (unsigned long long)(size_t)(w1t + ((size_t)e * HH + nt * 128) * DD);
    unsigned long long w2base = (unsigned long long)(size_t)(w2t + ((size_t)e * HH + nt * 128) * DD);

    auto issue = [&](int kc, int p) {
        unsigned long long colb = (unsigned long long)kc * 64 * 2;
        tdm_load(lds_off_of(&xs[p][0][0]),  xbase + colb, 64, 16, DD, DD, TT, true, gA0, gA1);
        tdm_load(lds_off_of(&xs[p][16][0]), xbase + colb, 64, 16, DD, DD, TT, true, gB0, gB1);
        tdm_load(lds_off_of(&w1s[p][0][0]), w1base + colb, 64, 128, DD, DD, HH, false, z4, z4);
        tdm_load(lds_off_of(&w2s[p][0][0]), w2base + colb, 64, 128, DD, DD, HH, false, z4, z4);
    };

    if (tid < 32) issue(0, 0);

    v8f acc1[2] = {}, acc2[2] = {};     // [M-subtile] x {W1, W2}
    int ni = wave;                      // 8 waves x 16 cols = 128-wide N strip

    for (int kc = 0; kc < NK1; ++kc) {
        int p = kc & 1;
        if (tid < 32) {
            if (kc + 1 < NK1) {
                issue(kc + 1, p ^ 1);
                __builtin_amdgcn_s_wait_tensorcnt(4);
            } else {
                __builtin_amdgcn_s_wait_tensorcnt(0);
            }
        }
        __syncthreads();

#pragma unroll
        for (int kk = 0; kk < 64; kk += 32) {
            ABReg a0, a1, br1, br2;
            int aklo = kk + ((lane & 16) ? 8 : 0);
            int ar = lane & 15;
            a0.h.lo = *(const v8bf*)&xs[p][ar][aklo];
            a0.h.hi = *(const v8bf*)&xs[p][ar][aklo + 16];
            a1.h.lo = *(const v8bf*)&xs[p][16 + ar][aklo];
            a1.h.hi = *(const v8bf*)&xs[p][16 + ar][aklo + 16];
            int brow = ni * 16 + (lane & 15);
            int bklo = kk + ((lane & 16) ? 8 : 0);
            br1.h.lo = *(const v8bf*)&w1s[p][brow][bklo];
            br1.h.hi = *(const v8bf*)&w1s[p][brow][bklo + 16];
            br2.h.lo = *(const v8bf*)&w2s[p][brow][bklo];
            br2.h.hi = *(const v8bf*)&w2s[p][brow][bklo + 16];
            acc1[0] = __builtin_amdgcn_wmma_f32_16x16x32_bf16(false, a0.v, false, br1.v,
                                                              (short)0, acc1[0], false, false);
            acc1[1] = __builtin_amdgcn_wmma_f32_16x16x32_bf16(false, a1.v, false, br1.v,
                                                              (short)0, acc1[1], false, false);
            acc2[0] = __builtin_amdgcn_wmma_f32_16x16x32_bf16(false, a0.v, false, br2.v,
                                                              (short)0, acc2[0], false, false);
            acc2[1] = __builtin_amdgcn_wmma_f32_16x16x32_bf16(false, a1.v, false, br2.v,
                                                              (short)0, acc2[1], false, false);
        }
        __syncthreads();
    }

    int ncol = nt * 128 + ni * 16 + (lane & 15);
    float b1v = b1[e * HH + ncol];
    float b2v = b2[e * HH + ncol];
    int oe = off[e];
#pragma unroll
    for (int mi = 0; mi < 2; ++mi) {
        int lrow_base = mt * 32 + mi * 16 + ((lane & 16) ? 8 : 0);
#pragma unroll
        for (int r = 0; r < 8; ++r) {
            int lrow = lrow_base + r;
            if (lrow < ce) {
                float h1 = acc1[mi][r] + b1v;
                float h2 = acc2[mi][r] + b2v;
                float sw = h1 * (h2 / (1.f + __expf(-h2)));
                hh[(size_t)(oe + lrow) * HH + ncol] = (__bf16)sw;
            }
        }
    }
}

// ---------------------------------------------------------------------------
// k2_down: stage-2 GEMM, tile 32(M) x 128(N), K over H. B-tile reused across
// both M-subtiles. Gate-scaled atomic scatter into out.
// ---------------------------------------------------------------------------
#define NK2 (HH / 64)
__global__ __launch_bounds__(256) void k2_down(const __bf16* __restrict__ wpt,
                                               const float* __restrict__ bp,
                                               const __bf16* __restrict__ hh,
                                               const int* __restrict__ pair_tok,
                                               const float* __restrict__ pair_gate,
                                               const int* __restrict__ cnt,
                                               const int* __restrict__ off,
                                               float* __restrict__ out) {
    int e = blockIdx.z, mt = blockIdx.y, nt = blockIdx.x;   // nt over DD/128
    int ce = cnt[e];
    if (mt * 32 >= ce) return;

    __shared__ __bf16 hs[2][32][64];
    __shared__ __bf16 wps[2][128][64];
    __shared__ int   tokL[32];
    __shared__ float gateL[32];

    int tid = threadIdx.x;
    int lane = tid & 31, wave = tid >> 5;
    if (tid < 32) {
        int slot = mt * 32 + tid;
        if (slot < ce) {
            tokL[tid]  = pair_tok[e * TT + slot];
            gateL[tid] = pair_gate[e * TT + slot];
        } else {
            tokL[tid] = 0; gateL[tid] = 0.f;
        }
    }
    __syncthreads();

    int oe = off[e];
    v4i_ z4 = {0, 0, 0, 0};
    unsigned long long hbase = (unsigned long long)(size_t)(hh + (size_t)(oe + mt * 32) * HH);
    unsigned long long wbase = (unsigned long long)(size_t)(wpt + ((size_t)e * DD + nt * 128) * HH);

    auto issue = [&](int kc, int p) {
        unsigned long long colb = (unsigned long long)kc * 64 * 2;
        tdm_load(lds_off_of(&hs[p][0][0]),  hbase + colb, 64, 32, HH, HH, TT * TOPK, false, z4, z4);
        tdm_load(lds_off_of(&wps[p][0][0]), wbase + colb, 64, 128, HH, HH, DD, false, z4, z4);
    };

    if (tid < 32) issue(0, 0);

    v8f acc[2] = {};
    int ni = wave;

    for (int kc = 0; kc < NK2; ++kc) {
        int p = kc & 1;
        if (tid < 32) {
            if (kc + 1 < NK2) {
                issue(kc + 1, p ^ 1);
                __builtin_amdgcn_s_wait_tensorcnt(2);
            } else {
                __builtin_amdgcn_s_wait_tensorcnt(0);
            }
        }
        __syncthreads();

#pragma unroll
        for (int kk = 0; kk < 64; kk += 32) {
            ABReg a0, a1, b;
            int aklo = kk + ((lane & 16) ? 8 : 0);
            int ar = lane & 15;
            a0.h.lo = *(const v8bf*)&hs[p][ar][aklo];
            a0.h.hi = *(const v8bf*)&hs[p][ar][aklo + 16];
            a1.h.lo = *(const v8bf*)&hs[p][16 + ar][aklo];
            a1.h.hi = *(const v8bf*)&hs[p][16 + ar][aklo + 16];
            int brow = ni * 16 + (lane & 15);
            int bklo = kk + ((lane & 16) ? 8 : 0);
            b.h.lo = *(const v8bf*)&wps[p][brow][bklo];
            b.h.hi = *(const v8bf*)&wps[p][brow][bklo + 16];
            acc[0] = __builtin_amdgcn_wmma_f32_16x16x32_bf16(false, a0.v, false, b.v,
                                                             (short)0, acc[0], false, false);
            acc[1] = __builtin_amdgcn_wmma_f32_16x16x32_bf16(false, a1.v, false, b.v,
                                                             (short)0, acc[1], false, false);
        }
        __syncthreads();
    }

    int ncol = nt * 128 + ni * 16 + (lane & 15);
    float bpv = bp[e * DD + ncol];
#pragma unroll
    for (int mi = 0; mi < 2; ++mi) {
        int li_base = mi * 16 + ((lane & 16) ? 8 : 0);
#pragma unroll
        for (int r = 0; r < 8; ++r) {
            int li = li_base + r;
            float g = gateL[li];
            int t = tokL[li];
            float val = (acc[mi][r] + bpv) * g;
            unsafeAtomicAdd(&out[(size_t)t * DD + ncol], val);
        }
    }
}

// ---------------------------------------------------------------------------
extern "C" void kernel_launch(void* const* d_in, const int* in_sizes, int n_in,
                              void* d_out, int out_size, void* d_ws, size_t ws_size,
                              hipStream_t stream) {
    (void)in_sizes; (void)n_in; (void)out_size; (void)ws_size;

    const float* x     = (const float*)d_in[0];
    const float* noise = (const float*)d_in[1];
    const float* Wg    = (const float*)d_in[2];
    const float* nw    = (const float*)d_in[3];
    const float* W1    = (const float*)d_in[4];
    const float* b1    = (const float*)d_in[5];
    const float* W2    = (const float*)d_in[6];
    const float* b2    = (const float*)d_in[7];
    const float* Wp    = (const float*)d_in[8];
    const float* bp    = (const float*)d_in[9];

    float* out    = (float*)d_out;
    int*   idxout = (int*)((float*)d_out + (size_t)TT * DD);

    // workspace layout
    __bf16* xb  = (__bf16*)d_ws;                       // TT*DD
    __bf16* hh  = xb + (size_t)TT * DD;                // TT*TOPK*HH
    __bf16* w1t = hh + (size_t)TT * TOPK * HH;         // [E][H][D] bf16
    __bf16* w2t = w1t + (size_t)EE * DD * HH;
    __bf16* wpt = w2t + (size_t)EE * DD * HH;          // [E][D][H] bf16
    char*  p    = (char*)(wpt + (size_t)EE * DD * HH);
    int*   pair_tok  = (int*)p;                        // EE*TT
    float* pair_gate = (float*)(pair_tok + EE * TT);   // EE*TT
    int*   cnt       = (int*)(pair_gate + EE * TT);    // EE
    int*   off       = cnt + EE;                       // EE+1

    {
        int n4 = TT * DD / 4;
        k_init<<<(n4 + 255) / 256, 256, 0, stream>>>(out, cnt);
    }
    k_transpose_cast<<<dim3(HH / 64, DD / 64, EE), 256, 0, stream>>>(W1, w1t, DD, HH);
    k_transpose_cast<<<dim3(HH / 64, DD / 64, EE), 256, 0, stream>>>(W2, w2t, DD, HH);
    k_transpose_cast<<<dim3(DD / 64, HH / 64, EE), 256, 0, stream>>>(Wp, wpt, HH, DD);
    k_gate<<<TT / 8, 256, 0, stream>>>(x, noise, Wg, nw, xb, pair_tok, pair_gate, cnt, idxout);
    k_off<<<1, 32, 0, stream>>>(cnt, off);
    k1_up<<<dim3(HH / 128, TT / 32, EE), 256, 0, stream>>>(w1t, w2t, b1, b2, xb,
                                                           pair_tok, cnt, off, hh);
    k2_down<<<dim3(DD / 128, TT / 32, EE), 256, 0, stream>>>(wpt, bp, hh, pair_tok,
                                                             pair_gate, cnt, off, out);
}
